// GCN_69441031242461
// MI455X (gfx1250) — matrix operand order repaired and checked by reading, not compiled, not run on hardware
//
#include <hip/hip_runtime.h>
#include <cstdint>

typedef __attribute__((ext_vector_type(16))) _Float16 v16h;
typedef __attribute__((ext_vector_type(8)))  float    v8f;

#define NEG_SLOPE 0.2f

// ---------------- utility kernels ----------------

__global__ __launch_bounds__(256) void k_f32_to_f16(const float* __restrict__ in,
                                                    _Float16* __restrict__ out, int n) {
  int t = blockIdx.x * blockDim.x + threadIdx.x;
  if (t < n) out[t] = (_Float16)in[t];
}

// W: [K x C] row-major f32  ->  Wt: [C x K] row-major f16 (so GEMM B-fragments
// load K-contiguous pairs as single dwords).
__global__ __launch_bounds__(256) void k_transpose_f16(const float* __restrict__ W,
                                                       _Float16* __restrict__ Wt,
                                                       int K, int C) {
  int t = blockIdx.x * blockDim.x + threadIdx.x;
  if (t >= K * C) return;
  int k = t / C, c = t - k * C;
  Wt[c * K + k] = (_Float16)W[t];
}

__global__ __launch_bounds__(256) void k_fill(float* __restrict__ p, float v, int n) {
  int t = blockIdx.x * blockDim.x + threadIdx.x;
  if (t < n) p[t] = v;
}

// ---------------- WMMA GEMM: out[N x C] = A[N x 128](f16) * W[128 x C] ----------------
// W passed pre-transposed as Wt[C x 128] f16.  One wave32 computes a 16x16 tile,
// accumulating K=128 in four v_wmma_f32_16x16x32_f16 steps.  Requires N % 16 == 0
// (true here: 100000 = 6250*16) and C % 16 == 0.
__global__ __launch_bounds__(128)
void k_wmma_gemm(const _Float16* __restrict__ A,
                 const _Float16* __restrict__ Wt,
                 float* __restrict__ out, int C) {
  const int lane    = threadIdx.x & 31;
  const int wave    = threadIdx.x >> 5;
  const int rowBase = blockIdx.x * 16;
  const int colBase = (blockIdx.y * (blockDim.x >> 5) + wave) * 16;
  if (colBase >= C) return;                 // uniform per wave
  const int halfSel = (lane >> 4) & 1;      // 0: lanes 0-15, 1: lanes 16-31
  const int l15     = lane & 15;

  const _Float16* arow = A  + (size_t)(rowBase + l15) * 128;   // A row for M=l15
  const _Float16* brow = Wt + (size_t)(colBase + l15) * 128;   // Wt row for N=l15

  v8f c = {};
  for (int kk = 0; kk < 128; kk += 32) {
    union { v16h v; unsigned u[8]; } a, b;
#pragma unroll
    for (int d = 0; d < 8; ++d) {
      // A 16x32 f16 layout: elem pairs (2d,2d+1); d<4 -> K=2d, d>=4 -> K=2d+8;
      // lanes 16-31 shift K by +8 within each 16-wide K half.
      int ka = kk + 2 * d + ((d & 4) << 1) + halfSel * 8;
      a.u[d] = *(const unsigned*)(arow + ka);
      // B 32x16 f16 layout: same K mapping as A, N = lane&15.
      int kb = kk + 2 * d + ((d & 4) << 1) + halfSel * 8;
      b.u[d] = *(const unsigned*)(brow + kb);
    }
    c = __builtin_amdgcn_wmma_f32_16x16x32_f16(
        /*neg_a=*/false, a.v, /*neg_b=*/false, b.v,
        /*c_mod=*/(short)0, c, /*reuse_a=*/false, /*reuse_b=*/false);
  }
  // C/D layout: VGPR r -> M = r + 8*halfSel, N = lane&15
  float* ocol = out + (size_t)rowBase * C + colBase + l15;
#pragma unroll
  for (int r = 0; r < 8; ++r)
    ocol[(size_t)(r + 8 * halfSel) * C] = c[r];
}

// ---------------- edge-phase kernels ----------------

__device__ inline void atomicMaxFloat(float* addr, float val) {
  unsigned* a = (unsigned*)addr;
  unsigned old = *a;
  while (__uint_as_float(old) < val) {
    unsigned assumed = old;
    old = atomicCAS(a, assumed, __float_as_uint(val));
    if (old == assumed) break;
  }
}

// Group of C/4 lanes per edge: e = att . leaky_relu(xl[src] + xr[dst]).
// Each lane loads one float4 of the feature rows (single b128, fully coalesced),
// reduces within the group via wave32 shuffles, tracks segment max.
template <int C>
__global__ __launch_bounds__(256)
void k_edge_score(const long long* __restrict__ src, const long long* __restrict__ dst,
                  int nE, int nN,
                  const float* __restrict__ xl, const float* __restrict__ xr,
                  const float* __restrict__ att,
                  float* __restrict__ eBuf, float* __restrict__ nodeMax) {
  constexpr int GS = C / 4;                         // 32 (C=128) or 16 (C=64)
  int grp  = threadIdx.x / GS;
  int li   = threadIdx.x % GS;
  int edge = blockIdx.x * (256 / GS) + grp;
  if (edge >= nE + nN) return;
  long long s, d;
  if (edge < nE) { s = src[edge]; d = dst[edge]; }
  else           { s = d = edge - nE; }             // self loop
  const float4 l = ((const float4*)(xl + (size_t)s * C))[li];
  const float4 r = ((const float4*)(xr + (size_t)d * C))[li];
  const float4 a = ((const float4*)att)[li];
  float h0 = l.x + r.x, h1 = l.y + r.y, h2 = l.z + r.z, h3 = l.w + r.w;
  h0 = (h0 > 0.f) ? h0 : NEG_SLOPE * h0;
  h1 = (h1 > 0.f) ? h1 : NEG_SLOPE * h1;
  h2 = (h2 > 0.f) ? h2 : NEG_SLOPE * h2;
  h3 = (h3 > 0.f) ? h3 : NEG_SLOPE * h3;
  float acc = a.x * h0 + a.y * h1 + a.z * h2 + a.w * h3;
#pragma unroll
  for (int o = GS / 2; o > 0; o >>= 1) acc += __shfl_xor(acc, o, GS);
  if (li == 0) {
    eBuf[edge] = acc;
    atomicMaxFloat(&nodeMax[d], acc);
  }
}

__global__ __launch_bounds__(256)
void k_edge_expsum(const long long* __restrict__ dst, int nE, int nN,
                   float* __restrict__ eBuf, const float* __restrict__ nodeMax,
                   float* __restrict__ denom) {
  int t = blockIdx.x * blockDim.x + threadIdx.x;
  if (t >= nE + nN) return;
  long long d = (t < nE) ? dst[t] : (long long)(t - nE);
  float ex = __expf(eBuf[t] - nodeMax[d]);
  eBuf[t] = ex;
  atomicAdd(&denom[d], ex);
}

__global__ __launch_bounds__(256)
void k_edge_alpha(const long long* __restrict__ dst, int nE, int nN,
                  float* __restrict__ eBuf, const float* __restrict__ denom) {
  int t = blockIdx.x * blockDim.x + threadIdx.x;
  if (t >= nE + nN) return;
  long long d = (t < nE) ? dst[t] : (long long)(t - nE);
  eBuf[t] = eBuf[t] / denom[d];
}

// Group of C/4 lanes per edge: acc[dst] += alpha * xl[src].
// float4 gather of the source row, four f32 atomics per lane; indices/alpha
// loaded once per group instead of once per channel.
template <int C>
__global__ __launch_bounds__(256)
void k_edge_aggregate(const long long* __restrict__ src, const long long* __restrict__ dst,
                      int nE, int nN,
                      const float* __restrict__ eBuf, const float* __restrict__ xl,
                      float* __restrict__ acc) {
  constexpr int GS = C / 4;
  int grp  = threadIdx.x / GS;
  int li   = threadIdx.x % GS;
  int edge = blockIdx.x * (256 / GS) + grp;
  if (edge >= nE + nN) return;
  long long s, d;
  if (edge < nE) { s = src[edge]; d = dst[edge]; }
  else           { s = d = edge - nE; }
  float alpha = eBuf[edge];
  float4 v = ((const float4*)(xl + (size_t)s * C))[li];
  float* p = acc + (size_t)d * C + li * 4;
  atomicAdd(p + 0, alpha * v.x);
  atomicAdd(p + 1, alpha * v.y);
  atomicAdd(p + 2, alpha * v.z);
  atomicAdd(p + 3, alpha * v.w);
}

// h = relu(acc + bias) -> f16 (feeds layer-2 WMMA GEMM)
template <int C>
__global__ __launch_bounds__(256)
void k_bias_relu_f16(const float* __restrict__ acc, const float* __restrict__ bias,
                     _Float16* __restrict__ outh, int total) {
  int t = blockIdx.x * blockDim.x + threadIdx.x;
  if (t >= total) return;
  float v = acc[t] + bias[t & (C - 1)];
  outh[t] = (_Float16)fmaxf(v, 0.f);
}

template <int C>
__global__ __launch_bounds__(256)
void k_bias_add(float* __restrict__ out, const float* __restrict__ bias, int total) {
  int t = blockIdx.x * blockDim.x + threadIdx.x;
  if (t >= total) return;
  out[t] += bias[t & (C - 1)];
}

// ---------------- host-side orchestration ----------------

static inline int cdiv(long long a, long long b) { return (int)((a + b - 1) / b); }

extern "C" void kernel_launch(void* const* d_in, const int* in_sizes, int n_in,
                              void* d_out, int out_size, void* d_ws, size_t ws_size,
                              hipStream_t stream) {
  const float*     x    = (const float*)d_in[0];
  const long long* ei   = (const long long*)d_in[1];   // int64 [2, E]
  const float*     Wl1  = (const float*)d_in[2];
  const float*     Wr1  = (const float*)d_in[3];
  const float*     att1 = (const float*)d_in[4];
  const float*     b1   = (const float*)d_in[5];
  const float*     Wl2  = (const float*)d_in[6];
  const float*     Wr2  = (const float*)d_in[7];
  const float*     att2 = (const float*)d_in[8];
  const float*     b2   = (const float*)d_in[9];
  float*           out  = (float*)d_out;

  const int N = in_sizes[0] / 128;       // 100000 (divisible by 16)
  const int E = in_sizes[1] / 2;         // 1600000
  const int T = E + N;                   // edges incl. self loops
  const long long* srcIdx = ei;
  const long long* dstIdx = ei + E;

  // workspace carve-out (~187 MB)
  char* ws = (char*)d_ws;
  size_t off = 0;
  auto carve = [&](size_t bytes) -> void* {
    void* p = ws + off;
    off = (off + bytes + 255) & ~(size_t)255;
    return p;
  };
  _Float16* xh    = (_Float16*)carve((size_t)N * 128 * 2); // f16 activations (x, later h1)
  float*    xl    = (float*)carve((size_t)N * 128 * 4);
  float*    xr    = (float*)carve((size_t)N * 128 * 4);
  float*    acc   = (float*)carve((size_t)N * 128 * 4);    // layer-1 aggregation
  float*    eBuf  = (float*)carve((size_t)T * 4);
  float*    nodeM = (float*)carve((size_t)N * 4);
  float*    nodeD = (float*)carve((size_t)N * 4);
  _Float16* wtA   = (_Float16*)carve(128 * 128 * 2);       // W^T f16
  _Float16* wtB   = (_Float16*)carve(128 * 128 * 2);
  (void)ws_size; (void)n_in; (void)out_size;

  const float NEG_INF = -3.0e38f;

  // ================= Layer 1 (C = 128) =================
  k_f32_to_f16<<<cdiv((long long)N * 128, 256), 256, 0, stream>>>(x, xh, N * 128);
  k_transpose_f16<<<cdiv(128 * 128, 256), 256, 0, stream>>>(Wl1, wtA, 128, 128);
  k_transpose_f16<<<cdiv(128 * 128, 256), 256, 0, stream>>>(Wr1, wtB, 128, 128);

  dim3 gg1(N / 16, 2);   // 8 col tiles / 4 waves per block
  k_wmma_gemm<<<gg1, 128, 0, stream>>>(xh, wtA, xl, 128);
  k_wmma_gemm<<<gg1, 128, 0, stream>>>(xh, wtB, xr, 128);

  k_fill<<<cdiv(N, 256), 256, 0, stream>>>(nodeM, NEG_INF, N);
  k_fill<<<cdiv(N, 256), 256, 0, stream>>>(nodeD, 0.f, N);
  k_fill<<<cdiv((long long)N * 128, 256), 256, 0, stream>>>(acc, 0.f, N * 128);

  k_edge_score<128><<<cdiv(T, 8), 256, 0, stream>>>(srcIdx, dstIdx, E, N, xl, xr,
                                                    att1, eBuf, nodeM);
  k_edge_expsum<<<cdiv(T, 256), 256, 0, stream>>>(dstIdx, E, N, eBuf, nodeM, nodeD);
  k_edge_alpha<<<cdiv(T, 256), 256, 0, stream>>>(dstIdx, E, N, eBuf, nodeD);
  k_edge_aggregate<128><<<cdiv(T, 8), 256, 0, stream>>>(
      srcIdx, dstIdx, E, N, eBuf, xl, acc);

  // h1 = relu(acc + b1) -> f16 into xh (input of layer 2 GEMMs)
  k_bias_relu_f16<128><<<cdiv((long long)N * 128, 256), 256, 0, stream>>>(
      acc, b1, xh, N * 128);

  // ================= Layer 2 (C = 64) =================
  k_transpose_f16<<<cdiv(128 * 64, 256), 256, 0, stream>>>(Wl2, wtA, 128, 64);
  k_transpose_f16<<<cdiv(128 * 64, 256), 256, 0, stream>>>(Wr2, wtB, 128, 64);

  dim3 gg2(N / 16, 1);   // 4 col tiles / 4 waves per block
  k_wmma_gemm<<<gg2, 128, 0, stream>>>(xh, wtA, xl, 64);
  k_wmma_gemm<<<gg2, 128, 0, stream>>>(xh, wtB, xr, 64);

  k_fill<<<cdiv(N, 256), 256, 0, stream>>>(nodeM, NEG_INF, N);
  k_fill<<<cdiv(N, 256), 256, 0, stream>>>(nodeD, 0.f, N);
  k_fill<<<cdiv((long long)N * 64, 256), 256, 0, stream>>>(out, 0.f, N * 64);

  k_edge_score<64><<<cdiv(T, 16), 256, 0, stream>>>(srcIdx, dstIdx, E, N, xl, xr,
                                                    att2, eBuf, nodeM);
  k_edge_expsum<<<cdiv(T, 256), 256, 0, stream>>>(dstIdx, E, N, eBuf, nodeM, nodeD);
  k_edge_alpha<<<cdiv(T, 256), 256, 0, stream>>>(dstIdx, E, N, eBuf, nodeD);
  k_edge_aggregate<64><<<cdiv(T, 16), 256, 0, stream>>>(
      srcIdx, dstIdx, E, N, eBuf, xl, out);

  k_bias_add<64><<<cdiv((long long)N * 64, 256), 256, 0, stream>>>(out, b2, N * 64);
}
